// PoincareConvolution2d_9680856286019
// MI455X (gfx1250) — matrix operand order repaired and checked by reading, not compiled, not run on hardware
//
#include <hip/hip_runtime.h>
#include <math.h>

// ---------------- problem constants ----------------
#define C_INC   256
#define C_OUTC  256
#define BATCH   32
#define HWDIM   32
#define HPAD    34              // padded spatial (32 + 2)
#define KCH     72              // K chunks of 32 (2304 / 32)
#define LWP     257             // padded LDS row stride (bank-conflict-free)

typedef __attribute__((ext_vector_type(8)))  _Float16 v8h;
typedef __attribute__((ext_vector_type(16))) _Float16 v16h;
typedef __attribute__((ext_vector_type(8)))  float    v8f;

// ---------------- workspace layout (bytes) ----------------
#define XH_OFF   0
#define XH_BYTES (BATCH*HPAD*HPAD*C_INC*2)        // 18,939,904  padded NHWC fp16
#define SP_OFF   (XH_OFF + XH_BYTES)
#define SP_BYTES (BATCH*HPAD*HPAD*4)              // 147,968     padded per-pixel sum(x^2)
#define CO_OFF   (SP_OFF + SP_BYTES)
#define CO_BYTES (C_OUTC*4*4)                     // 4096        per-o consts
#define PB_OFF   (CO_OFF + CO_BYTES)
#define PB_BYTES (16*KCH*32*16*2)                 // 1,179,648   packed B fragments

// =====================================================================
// Kernel 1: x -> scaled, padded NHWC fp16  +  per-pixel channel sumsq
// grid = B*H (1024), block = 256 (w:32 x cgroup:8)
// =====================================================================
__global__ __launch_bounds__(256) void k_prep_x(const float* __restrict__ x,
                                                _Float16* __restrict__ xh,
                                                float* __restrict__ sp,
                                                float scale) {
    int bh = blockIdx.x;
    int b  = bh >> 5, h = bh & 31;
    int w  = threadIdx.x & 31;
    int c8 = threadIdx.x >> 5;           // 0..7
    float acc = 0.f;
    const float* xin = x + (((size_t)b * C_INC) * HWDIM + h) * HWDIM + w;
    _Float16* xo = xh + (((size_t)b * HPAD + (h + 1)) * HPAD + (w + 1)) * C_INC;
    #pragma unroll 4
    for (int i = 0; i < 32; ++i) {
        int c = c8 * 32 + i;
        float v = xin[(size_t)c * (HWDIM * HWDIM)] * scale;   // coalesced along w
        xo[c] = (_Float16)v;
        acc += v * v;
    }
    __shared__ float red[256];
    red[threadIdx.x] = acc;
    __syncthreads();
    if (threadIdx.x < 32) {
        float s = 0.f;
        #pragma unroll
        for (int g = 0; g < 8; ++g) s += red[g * 32 + w];
        sp[((size_t)b * HPAD + (h + 1)) * HPAD + (w + 1)] = s;
    }
}

// =====================================================================
// Kernel 2: per-output-column consts: 2*zn, cosh(2r), sinh(2r), 1/zn
// grid = 256 blocks x 256 threads (D = 2304 = 9*256)
// =====================================================================
__global__ __launch_bounds__(256) void k_prep_o(const float* __restrict__ wt,
                                                const float* __restrict__ bias,
                                                float* __restrict__ co) {
    int o = blockIdx.x, t = threadIdx.x;
    float acc = 0.f;
    #pragma unroll
    for (int i = 0; i < 9; ++i) {
        float v = wt[(size_t)(i * 256 + t) * C_OUTC + o];
        acc += v * v;
    }
    __shared__ float red[256];
    red[t] = acc; __syncthreads();
    for (int s = 128; s > 0; s >>= 1) { if (t < s) red[t] += red[t + s]; __syncthreads(); }
    if (t == 0) {
        float zn = fmaxf(sqrtf(red[0]), 1e-15f);
        float tb = 2.f * bias[o];
        co[o]             = 2.f * zn;
        co[C_OUTC + o]    = coshf(tb);
        co[2*C_OUTC + o]  = sinhf(tb);
        co[3*C_OUTC + o]  = 1.f / zn;
    }
}

// =====================================================================
// Kernel 3: pack z/zn into WMMA B-fragment order, K permuted to (ij,c)
// blk = nt*KCH + kc ; lane layout: n = nt*16 + lane%16 ; k = 8*(lane/16)+t+(t&8)
// grid = 16*72 blocks x 32 threads
// =====================================================================
__global__ __launch_bounds__(32) void k_pack_b(const float* __restrict__ wt,
                                               const float* __restrict__ co,
                                               _Float16* __restrict__ pb) {
    int blk  = blockIdx.x;
    int nt   = blk / KCH, kc = blk % KCH;
    int lane = threadIdx.x;
    int n    = nt * 16 + (lane & 15);
    float inv = co[3*C_OUTC + n];
    int ij = kc >> 3, cbase = (kc & 7) * 32;
    int kbase = (lane >> 4) * 8;
    _Float16* out = pb + ((size_t)blk * 32 + lane) * 16;
    #pragma unroll
    for (int t = 0; t < 16; ++t) {
        int kl = kbase + t + (t & 8);       // 0..31 within chunk
        int c  = cbase + kl;                // permuted K: k' = ij*256 + c
        int d  = c * 9 + ij;                // original unfold index
        out[t] = (_Float16)(wt[(size_t)d * C_OUTC + n] * inv);
    }
}

// =====================================================================
// Kernel 4: fused implicit-im2col WMMA GEMM + hyperbolic epilogue
// grid = 1024 workgroups (32 pixels each) x 128 threads (4 waves)
// All waves share the 32 rows; wave i owns columns [64i, 64i+64).
// Per K-chunk: 2 A fragments (rows 0-15 / 16-31) reuse 4 B fragments.
// Epilogue stages w in LDS (kills acc liveness -> no scratch spills)
// and finishes with a coalesced, row-mapped output pass.
// =====================================================================
__global__ __launch_bounds__(128) void k_gemm(const _Float16* __restrict__ xh,
                                              const float* __restrict__ sp,
                                              const float* __restrict__ co,
                                              const _Float16* __restrict__ pb,
                                              float* __restrict__ out) {
    int wg   = blockIdx.x;
    int tid  = threadIdx.x;
    int wave = tid >> 5, lane = tid & 31;

    // A-row pixels for this lane (A layout: row m = lane%16)
    int m  = lane & 15;
    int p0 = wg * 32 + m;            // rows 0..15 of the tile
    int p1 = p0 + 16;                // rows 16..31
    int b0 = p0 >> 10, h0 = (p0 >> 5) & 31, w0 = p0 & 31;
    int b1 = p1 >> 10, h1 = (p1 >> 5) & 31, w1 = p1 & 31;
    const _Float16* abase0 = xh + (((size_t)b0 * HPAD + h0) * HPAD + w0) * C_INC;
    const _Float16* abase1 = xh + (((size_t)b1 * HPAD + h1) * HPAD + w1) * C_INC;
    int hi8 = (lane >> 4) * 8;       // A interleave: lanes 16-31 start at K+8

    v8f acc[2][4];
    #pragma unroll
    for (int f = 0; f < 2; ++f)
        #pragma unroll
        for (int j = 0; j < 4; ++j) acc[f][j] = v8f{};

    const v16h* bcol = (const v16h*)pb + (size_t)(wave * 4) * KCH * 32 + lane;

    #pragma unroll 2
    for (int kc = 0; kc < KCH; ++kc) {
        int ij = kc >> 3;
        int cbase = (kc & 7) << 5;
        int di = ij / 3, dj = ij - di * 3;
        size_t aoff = ((size_t)di * HPAD + dj) * C_INC + cbase + hi8;

        const _Float16* ap0 = abase0 + aoff;
        v8h a00 = *(const v8h*)ap0;            // K = hi8 .. hi8+7
        v8h a01 = *(const v8h*)(ap0 + 16);     // K = hi8+16 .. hi8+23
        v16h a0 = __builtin_shufflevector(a00, a01, 0,1,2,3,4,5,6,7,8,9,10,11,12,13,14,15);

        const _Float16* ap1 = abase1 + aoff;
        v8h a10 = *(const v8h*)ap1;
        v8h a11 = *(const v8h*)(ap1 + 16);
        v16h a1 = __builtin_shufflevector(a10, a11, 0,1,2,3,4,5,6,7,8,9,10,11,12,13,14,15);

        const v16h* bp = bcol + (size_t)kc * 32;
        v16h bf0 = bp[0];
        v16h bf1 = bp[(size_t)KCH * 32];
        v16h bf2 = bp[(size_t)2 * KCH * 32];
        v16h bf3 = bp[(size_t)3 * KCH * 32];

        acc[0][0] = __builtin_amdgcn_wmma_f32_16x16x32_f16(false, a0, false, bf0, (short)0, acc[0][0], false, false);
        acc[0][1] = __builtin_amdgcn_wmma_f32_16x16x32_f16(false, a0, false, bf1, (short)0, acc[0][1], false, false);
        acc[0][2] = __builtin_amdgcn_wmma_f32_16x16x32_f16(false, a0, false, bf2, (short)0, acc[0][2], false, false);
        acc[0][3] = __builtin_amdgcn_wmma_f32_16x16x32_f16(false, a0, false, bf3, (short)0, acc[0][3], false, false);
        acc[1][0] = __builtin_amdgcn_wmma_f32_16x16x32_f16(false, a1, false, bf0, (short)0, acc[1][0], false, false);
        acc[1][1] = __builtin_amdgcn_wmma_f32_16x16x32_f16(false, a1, false, bf1, (short)0, acc[1][1], false, false);
        acc[1][2] = __builtin_amdgcn_wmma_f32_16x16x32_f16(false, a1, false, bf2, (short)0, acc[1][2], false, false);
        acc[1][3] = __builtin_amdgcn_wmma_f32_16x16x32_f16(false, a1, false, bf3, (short)0, acc[1][3], false, false);
    }

    // ---- epilogue ----
    __shared__ float rowFac[32], rowLam[32], rowS[32];
    __shared__ float lw[32 * LWP];            // w matrix [row][o], padded stride

    if (tid < 32) {
        int pp = wg * 32 + tid;
        int bb = pp >> 10, hh = (pp >> 5) & 31, ww = pp & 31;
        const float* s0 = sp + ((size_t)bb * HPAD + hh) * HPAD + ww;
        float n2 = 0.f;
        #pragma unroll
        for (int di = 0; di < 3; ++di)
            #pragma unroll
            for (int dj = 0; dj < 3; ++dj)
                n2 += s0[di * HPAD + dj];
        float n  = sqrtf(fmaxf(n2, 1e-30f));
        float tn = tanhf(n);
        rowFac[tid] = tn / n;                     // tanh(n)/n
        rowLam[tid] = 2.f / (1.f - tn * tn);      // conformal factor lambda
        rowS[tid]   = 0.f;
    }
    __syncthreads();

    // C/D layout: lane -> col n0 = lane%16 (per tile), rows rb..rb+7 (+16 for frag 1)
    int rb = (lane >> 4) * 8;
    int n0 = lane & 15;
    float part[16];
    #pragma unroll
    for (int i = 0; i < 16; ++i) part[i] = 0.f;

    #pragma unroll
    for (int f = 0; f < 2; ++f) {
        #pragma unroll
        for (int j = 0; j < 4; ++j) {
            int o = wave * 64 + j * 16 + n0;
            float zn2 = co[o], ch = co[C_OUTC + o], sh = co[2*C_OUTC + o];
            #pragma unroll
            for (int mm = 0; mm < 8; ++mm) {
                int r = f * 16 + rb + mm;
                float g   = acc[f][j][mm];
                float lam = rowLam[r];
                float u   = lam * rowFac[r] * g * ch - (lam - 1.f) * sh;
                float v   = zn2 * asinhf(u);
                float wv  = sinhf(v);
                lw[r * LWP + o] = wv;             // stage in LDS; acc dies here
                part[f * 8 + mm] += wv * wv;
            }
        }
    }
    #pragma unroll
    for (int f = 0; f < 2; ++f)
        #pragma unroll
        for (int mm = 0; mm < 8; ++mm)
            atomicAdd(&rowS[f * 16 + rb + mm], part[f * 8 + mm]);   // ds_add_f32
    __syncthreads();

    // ---- coalesced finalize: thread -> (row = tid&31, o-group = tid>>5) ----
    {
        int r  = tid & 31;
        int og = tid >> 5;
        float S  = rowS[r];
        float q  = 1.f + sqrtf(1.f + S);          // Moebius denom
        float ny = sqrtf(fmaxf(S, 1e-30f)) / q;   // ||y||
        float ar = fminf(ny, 1.f - 1e-7f);
        float fc = atanhf(ar) / (ny * q);         // out = fc * w
        int pp   = wg * 32 + r;
        float* obase = out + ((size_t)(pp >> 10) * C_OUTC) * (HWDIM*HWDIM) + (pp & 1023);
        #pragma unroll 8
        for (int oo = 0; oo < 64; ++oo) {
            int o = og * 64 + oo;
            obase[(size_t)o * (HWDIM*HWDIM)] = fc * lw[r * LWP + o];
        }
    }
}

// =====================================================================
extern "C" void kernel_launch(void* const* d_in, const int* in_sizes, int n_in,
                              void* d_out, int out_size, void* d_ws, size_t ws_size,
                              hipStream_t stream) {
    const float* x    = (const float*)d_in[0];
    const float* wt   = (const float*)d_in[1];
    const float* bias = (const float*)d_in[2];
    float* out        = (float*)d_out;

    char* ws = (char*)d_ws;
    _Float16* xh = (_Float16*)(ws + XH_OFF);
    float*    sp = (float*)(ws + SP_OFF);
    float*    co = (float*)(ws + CO_OFF);
    _Float16* pb = (_Float16*)(ws + PB_OFF);

    // beta(D/2, 1/2) / beta(C_in/2, 1/2) via lgamma (host, deterministic)
    double lr = (lgamma(1152.0) - lgamma(1152.5)) - (lgamma(128.0) - lgamma(128.5));
    float scale = (float)exp(lr);

    // zero the padded fp16 tensor + padded sumsq (halo must be 0)
    hipMemsetAsync(ws, 0, (size_t)(XH_BYTES + SP_BYTES), stream);

    k_prep_x<<<BATCH * HWDIM, 256, 0, stream>>>(x, xh, sp, scale);
    k_prep_o<<<C_OUTC, 256, 0, stream>>>(wt, bias, co);
    k_pack_b<<<16 * KCH, 32, 0, stream>>>(wt, co, pb);
    k_gemm<<<(BATCH * HWDIM * HWDIM) / 32, 128, 0, stream>>>(xh, sp, co, pb, out);
}
